// PVRNNLayer_7258494730545
// MI455X (gfx1250) — matrix operand order repaired and checked by reading, not compiled, not run on hardware
//
#include <hip/hip_runtime.h>
#include <hip/hip_bf16.h>

// ---------------- problem constants ----------------
#define T_LEN  1024
#define D_SZ   512
#define Z_SZ   64
#define IN_SZ  512
#define B_SZ   32
#define KCAT   576          // D + Z (fused [d|z] @ [Wd|Wz].T)
#define DZ_STRIDE 592       // LDS row stride (bf16) with padding
#define TAU_LO 2.0f
#define TAU_HI 4.0f         // tau doubled for first D/2 channels
#define EPS_F  1e-6f
#define KLW    (1e-3f / 64.0f)        // W / Z  (W1 == W in reference)
#define NLLW   (1e-3f / 512.0f)       // WD1 / D
#define HALF_LOG_2PI 0.9189385332f

typedef __bf16 bf16;
typedef __attribute__((ext_vector_type(16))) __bf16 v16bf;
typedef __attribute__((ext_vector_type(8)))  float  v8f;

// ----- WMMA fragment loaders (wave32; A and B share contiguous k order, k = 16*(lane>>4) + i) -----
// A: 16x32 bf16 tile, row=lane&15, K {kb..kb+15}, kb=(lane>>4)*16
__device__ __forceinline__ v16bf load_a_bf16(const bf16* base, int stride, int lane) {
    const bf16* p = base + (lane & 15) * stride + ((lane >> 4) * 16);
    v16bf a;
#pragma unroll
    for (int i = 0; i < 16; ++i) a[i] = p[i];
    return a;
}
// B: 32x16 bf16 tile from row-major weight W[N][K]: lane&15 = column, (lane>>4)*16 = K half
__device__ __forceinline__ v16bf load_b_bf16(const bf16* W, int ldw, int nbase, int kbase, int lane) {
    const bf16* p = W + (size_t)(nbase + (lane & 15)) * ldw + kbase + ((lane >> 4) * 16);
    v16bf b;
#pragma unroll
    for (int i = 0; i < 16; ++i) b[i] = p[i];
    return b;
}

#define WMMA_BF16(A_, B_, C_) \
    __builtin_amdgcn_wmma_f32_16x16x32_bf16(false, (A_), false, (B_), (short)0, (C_), false, false)

// ---------------- kernel 1a: weight conversion to bf16 ----------------
// Wcat[n][k] = (k<512) ? Wd[n][k] : Wz[n][k-512]   (512 x 576)
// Wdzb       = bf16(Wdz)                           (128 x 512)
// Whdb       = bf16(Whd)                           (512 x 512)
__global__ __launch_bounds__(1024) void pvrnn_prep_weights(
    const float* __restrict__ Wd, const float* __restrict__ Wz,
    const float* __restrict__ Wdz, const float* __restrict__ Whd,
    bf16* __restrict__ Wcat, bf16* __restrict__ Wdzb, bf16* __restrict__ Whdb) {
    int idx = blockIdx.x * 1024 + threadIdx.x;
    const int NC = D_SZ * KCAT;                 // 294912
    const int ND = 2 * Z_SZ * D_SZ;             // 65536
    if (idx < NC) {
        int n = idx / KCAT, k = idx - n * KCAT;
        float v = (k < D_SZ) ? Wd[(size_t)n * D_SZ + k] : Wz[(size_t)n * Z_SZ + (k - D_SZ)];
        Wcat[idx] = (bf16)v;
    } else if (idx < NC + ND) {
        int j = idx - NC;
        Wdzb[j] = (bf16)Wdz[j];
    } else {
        int j = idx - NC - ND;
        if (j < D_SZ * IN_SZ) Whdb[j] = (bf16)Whd[j];
    }
}

// ---------------- kernel 1b: hd -> bf16 (one streaming pass, 16M elements) ----------------
__global__ __launch_bounds__(1024) void pvrnn_conv_hd(
    const float* __restrict__ hd, bf16* __restrict__ hdb) {
    size_t i = ((size_t)blockIdx.x * 1024 + threadIdx.x) * 8;
#pragma unroll
    for (int j = 0; j < 8; ++j) hdb[i + j] = (bf16)hd[i + j];
}

// ---------------- kernel 2: X = hd @ Whd.T + bias  (wide WMMA GEMM, all-bf16) -------------
// X is [B*T, D] = [32768, 512]. Each wave computes a 32x16 macro-tile (two m-tiles
// sharing one B fragment); 8 waves per block -> 4096 blocks.
__global__ __launch_bounds__(256) void pvrnn_gemm_x(
    const bf16* __restrict__ hdb, const bf16* __restrict__ Whdb,
    const float* __restrict__ bias, float* __restrict__ X) {
    int lane = threadIdx.x & 31;
    int wv   = threadIdx.x >> 5;
    int task = blockIdx.x * 8 + wv;             // 0 .. 32767
    int mp   = task >> 5;                       // 0 .. 1023  (pairs of 16-row tiles)
    int nt   = task & 31;                       // 0 .. 31
    const bf16* a0 = hdb + (size_t)(mp * 32) * IN_SZ;
    const bf16* a1 = a0 + (size_t)16 * IN_SZ;
    v8f c0 = {}, c1 = {};
#pragma unroll
    for (int kc = 0; kc < IN_SZ / 32; ++kc) {
        v16bf b  = load_b_bf16(Whdb, IN_SZ, nt * 16, kc * 32, lane);
        v16bf f0 = load_a_bf16(a0 + kc * 32, IN_SZ, lane);
        v16bf f1 = load_a_bf16(a1 + kc * 32, IN_SZ, lane);
        c0 = WMMA_BF16(f0, b, c0);
        c1 = WMMA_BF16(f1, b, c1);
    }
    int n = nt * 16 + (lane & 15);
    float bv = bias[n];
#pragma unroll
    for (int r = 0; r < 8; ++r) {
        int m0 = mp * 32 + (lane >> 4) * 8 + r;
        X[(size_t)m0 * D_SZ + n]        = c0[r] + bv;
        X[(size_t)(m0 + 16) * D_SZ + n] = c1[r] + bv;
    }
}

// ---------------- kernel 3: persistent single-WGP recurrence ----------------
__global__ __launch_bounds__(1024) void pvrnn_recurrent(
    const float* __restrict__ A, const float* __restrict__ noise,
    const float* __restrict__ Xws, const bf16* __restrict__ Wcat,
    const bf16* __restrict__ Wdzb, const float* __restrict__ init_h,
    const float* __restrict__ init_h_mu, float* __restrict__ h_ws,
    float* __restrict__ out) {
    __shared__ bf16  dz_sh[B_SZ][DZ_STRIDE];   // [d | z] state, bf16, 37 KB
    __shared__ float P_sh[B_SZ][2 * Z_SZ];     // prior pre-activations, 16 KB
    __shared__ float loss_sh;

    const int tid  = threadIdx.x;
    const int w    = tid >> 5;                  // wave 0..31
    const int lane = tid & 31;
    float loss_acc = 0.f;

    if (tid == 0) loss_sh = 0.f;

    // ---- init: h0 = init_h, d0 = tanh(h0), init-h NLL ----
    // element ownership identical to the GEMM2 epilogue mapping -> no cross-thread h hazards
#pragma unroll
    for (int s = 0; s < 2; ++s) {
        int mt = s;                             // tiles (mt, nt=w): mt 0/1 covers all 32 rows
        int n  = w * 16 + (lane & 15);
#pragma unroll
        for (int r = 0; r < 8; ++r) {
            int m = mt * 16 + (lane >> 4) * 8 + r;
            float h0 = init_h[(size_t)m * D_SZ + n];
            float df = h0 - init_h_mu[n];
            loss_acc += NLLW * (HALF_LOG_2PI + 0.5f * df * df);
            h_ws[(size_t)m * D_SZ + n] = h0;
            dz_sh[m][n] = (bf16)tanhf(h0);
        }
    }

    const int e0 = tid * 2;                     // 2048 (b,z) elements / 1024 threads
    float muqr[2], sgqr[2];

    for (int t = 0; t < T_LEN; ++t) {
        // ---- Phase A: posterior sample z = tanh(muA) + eps * exp(lnA) ----
#pragma unroll
        for (int i = 0; i < 2; ++i) {
            int e = e0 + i, b = e >> 6, zc = e & 63;
            const float* Ar = A + ((size_t)b * T_LEN + t) * (2 * Z_SZ);
            float mq = tanhf(Ar[zc]);
            float sq = expf(Ar[Z_SZ + zc]);
            float nz = noise[((size_t)b * T_LEN + t) * Z_SZ + zc];
            dz_sh[b][D_SZ + zc] = (bf16)(mq + nz * sq);
            muqr[i] = mq; sgqr[i] = sq;
        }
        if (t == 0) {                           // prior at t==0 is zeros
            float* pf = &P_sh[0][0];
#pragma unroll
            for (int j = 0; j < 4; ++j) pf[tid * 4 + j] = 0.f;
        }
        __syncthreads();

        // ---- Phase B1: WMMA compute, accumulators stay in VGPRs ----
        v8f acc0 = {}, acc1 = {}, acc2 = {};
        {
            // fused G = [d|z] @ [Wd|Wz].T ; wave w owns column tile nt=w for both m-tiles
            const bf16* a0 = &dz_sh[0][0];
            const bf16* a1 = &dz_sh[16][0];
            for (int kc = 0; kc < KCAT / 32; ++kc) {
                if (kc + 1 < KCAT / 32)         // gfx1250 global_prefetch path on weight stream
                    __builtin_prefetch(Wcat + (size_t)(w * 16 + (lane & 15)) * KCAT
                                            + (kc + 1) * 32 + ((lane >> 4) * 16), 0, 1);
                v16bf bfrag = load_b_bf16(Wcat, KCAT, w * 16, kc * 32, lane);
                v16bf af0   = load_a_bf16(a0 + kc * 32, DZ_STRIDE, lane);
                v16bf af1   = load_a_bf16(a1 + kc * 32, DZ_STRIDE, lane);
                acc0 = WMMA_BF16(af0, bfrag, acc0);
                acc1 = WMMA_BF16(af1, bfrag, acc1);
            }
        }
        if (w < 16 && t > 0) {
            // prior P = d @ Wdz.T : 16 tiles over waves 0..15 (mt=w>>3, nt=w&7)
            const bf16* a2 = &dz_sh[(w >> 3) * 16][0];
            for (int kc = 0; kc < D_SZ / 32; ++kc) {
                v16bf a = load_a_bf16(a2 + kc * 32, DZ_STRIDE, lane);
                v16bf b = load_b_bf16(Wdzb, D_SZ, (w & 7) * 16, kc * 32, lane);
                acc2 = WMMA_BF16(a, b, acc2);
            }
        }
        __syncthreads();

        // ---- Phase B2: epilogues (tau-gated h update, tanh, state writeback) ----
        {
            int n = w * 16 + (lane & 15);
            float taun = (n < D_SZ / 2) ? TAU_HI : TAU_LO;
            float itau = 1.0f / taun, dec = 1.0f - itau;
#pragma unroll
            for (int r = 0; r < 8; ++r) {
                int mr = (lane >> 4) * 8 + r;
                {   // m-tile 0
                    int m = mr;
                    size_t off = ((size_t)m * T_LEN + t) * D_SZ + n;
                    float hn = dec * h_ws[(size_t)m * D_SZ + n] + (acc0[r] + Xws[off]) * itau;
                    float dn = tanhf(hn);
                    h_ws[(size_t)m * D_SZ + n] = hn;
                    out[off] = dn;
                    dz_sh[m][n] = (bf16)dn;
                }
                {   // m-tile 1
                    int m = 16 + mr;
                    size_t off = ((size_t)m * T_LEN + t) * D_SZ + n;
                    float hn = dec * h_ws[(size_t)m * D_SZ + n] + (acc1[r] + Xws[off]) * itau;
                    float dn = tanhf(hn);
                    h_ws[(size_t)m * D_SZ + n] = hn;
                    out[off] = dn;
                    dz_sh[m][n] = (bf16)dn;
                }
            }
            if (w < 16 && t > 0) {
                int col = (w & 7) * 16 + (lane & 15);
                int mb  = (w >> 3) * 16;
#pragma unroll
                for (int r = 0; r < 8; ++r)
                    P_sh[mb + (lane >> 4) * 8 + r][col] = acc2[r];
            }
        }
        __syncthreads();

        // ---- Phase C: KL accumulation (posterior stats kept in registers) ----
#pragma unroll
        for (int i = 0; i < 2; ++i) {
            int e = e0 + i, b = e >> 6, zc = e & 63;
            float mp = tanhf(P_sh[b][zc]);
            float sp = expf(P_sh[b][Z_SZ + zc]);
            float dmu = muqr[i] - mp, sq = sgqr[i];
            loss_acc += KLW * (logf(sp + EPS_F) - logf(sq + EPS_F) - 0.5f
                               + 0.5f * (dmu * dmu + sq * sq) / (sp * sp + EPS_F));
        }
    }

    atomicAdd(&loss_sh, loss_acc);              // ds_add_f32 reduction
    __syncthreads();
    if (tid == 0) out[(size_t)B_SZ * T_LEN * D_SZ] = loss_sh;
}

// ---------------- host launch ----------------
extern "C" void kernel_launch(void* const* d_in, const int* in_sizes, int n_in,
                              void* d_out, int out_size, void* d_ws, size_t ws_size,
                              hipStream_t stream) {
    const float* hd        = (const float*)d_in[0];
    const float* A         = (const float*)d_in[1];
    const float* noise     = (const float*)d_in[2];
    const float* Wd        = (const float*)d_in[3];
    const float* Wz        = (const float*)d_in[4];
    const float* Wdz       = (const float*)d_in[5];
    const float* Whd       = (const float*)d_in[6];
    const float* bias_d    = (const float*)d_in[7];
    const float* init_h    = (const float*)d_in[8];
    const float* init_h_mu = (const float*)d_in[9];
    // d_in[10] = in_p0 (zeros) — handled implicitly by zeroing the prior at t==0
    float* out = (float*)d_out;

    // workspace layout (all offsets 256B aligned)
    char*  ws   = (char*)d_ws;
    float* Xws  = (float*)(ws);                        // 32768*512*4  = 67108864
    bf16*  hdb  = (bf16*)(ws + 67108864);              // 32768*512*2  = 33554432
    bf16*  Wcat = (bf16*)(ws + 100663296);             // 512*576*2    = 589824
    bf16*  Wdzb = (bf16*)(ws + 101253120);             // 128*512*2    = 131072
    bf16*  Whdb = (bf16*)(ws + 101384192);             // 512*512*2    = 524288
    float* h_ws = (float*)(ws + 101908480);            // 32*512*4     = 65536

    // 1a) weights -> bf16 (294912 + 65536 + 262144 = 622592 elements)
    pvrnn_prep_weights<<<608, 1024, 0, stream>>>(Wd, Wz, Wdz, Whd, Wcat, Wdzb, Whdb);
    // 1b) hd -> bf16 (16M elements, 8 per thread)
    pvrnn_conv_hd<<<2048, 1024, 0, stream>>>(hd, hdb);
    // 2) wide WMMA GEMM: X = hd @ Whd.T + bias (32768 wave-tasks, 8 per block)
    pvrnn_gemm_x<<<4096, 256, 0, stream>>>(hdb, Whdb, bias_d, Xws);
    // 3) persistent single-workgroup recurrence (32 waves on one WGP)
    pvrnn_recurrent<<<1, 1024, 0, stream>>>(A, noise, Xws, Wcat, Wdzb,
                                            init_h, init_h_mu, h_ws, out);
}
